// MultiHead_Attention_Lattice_rel_save_gpumm_31379031064642
// MI455X (gfx1250) — compile-verified, matrix operand order checked by reading, not run on hardware
//
#include <hip/hip_runtime.h>
#include <hip/hip_bf16.h>

// MultiHeadAttention with relative position bias, MI455X / gfx1250.
// B=4, L=1024, H=512, NH=8, D=64.
// Memory-bound: 134MB rel_pos stream dominates. All GEMMs via
// v_wmma_f32_16x16x32_f16 (f16 operands, f32 accum), flash-attention
// single pass so scores are never materialized.

#define B_  4
#define L_  1024
#define H_  512
#define NH_ 8
#define D_  64

typedef __attribute__((ext_vector_type(16))) _Float16 v16h;
typedef __attribute__((ext_vector_type(8)))  _Float16 v8h;
typedef __attribute__((ext_vector_type(8)))  float    v8f;

__device__ __forceinline__ v8f wmma16(v16h a, v16h b, v8f c) {
  return __builtin_amdgcn_wmma_f32_16x16x32_f16(
      /*neg_a=*/false, a, /*neg_b=*/false, b,
      /*c_mod=*/(short)0, c, /*reuse_a=*/false, /*reuse_b=*/false);
}

// Build a v16h operand from two 16-byte contiguous chunks.
// Per CDNA5 16-bit A/B layout: lanes 0-15 hold K={0..7,16..23},
// lanes 16-31 hold K={8..15,24..31}; caller passes the right chunk ptrs.
__device__ __forceinline__ v16h ld16(const _Float16* p0, const _Float16* p1) {
  v8h x = *(const v8h*)p0;
  v8h y = *(const v8h*)p1;
  v16h r;
#pragma unroll
  for (int i = 0; i < 8; ++i) { r[i] = x[i]; r[i + 8] = y[i]; }
  return r;
}

__device__ __forceinline__ float rowmax16(float v) {
#pragma unroll
  for (int m = 1; m < 16; m <<= 1) v = fmaxf(v, __shfl_xor(v, m, 32));
  return v;
}
__device__ __forceinline__ float rowsum16(float v) {
#pragma unroll
  for (int m = 1; m < 16; m <<= 1) v += __shfl_xor(v, m, 32);
  return v;
}

// ---------------------------------------------------------------- cvt
__global__ void cvt_f32_to_f16(const float* __restrict__ in,
                               _Float16* __restrict__ out, int n) {
  int i = blockIdx.x * blockDim.x + threadIdx.x;
  int stride = gridDim.x * blockDim.x;
  for (; i < n; i += stride) out[i] = (_Float16)in[i];
}

// ---------------------------------------------------------------- GEMM
// C[4096x512] = A[4096x512] @ Wt^T (row n of Wt = weights of out col n,
// torch Linear) + bias. Each wave computes a 16x64 tile: one A operand
// feeds 4 WMMAs per k-step (1.25 loads/WMMA, 4x less A traffic).
// mode 0: store f16, head-major [b][h][l][d]   (Q, K)
// mode 1: store f16, transposed  [b][h][d][l]  (V^T)
// mode 2: store f32 row-major [row][512]       (final output)
__global__ __launch_bounds__(128) void gemm512_kernel(
    const _Float16* __restrict__ A, const _Float16* __restrict__ Wt,
    const float* __restrict__ bias, void* __restrict__ out, int mode) {
  const int lane = threadIdx.x & 31;
  const int wave = threadIdx.x >> 5;
  const int half = lane >> 4;
  const int l15  = lane & 15;
  const int tile = blockIdx.x * 4 + wave;       // 2048 tiles total
  const int m0 = (tile >> 3) << 4;              // 256 row tiles
  const int n0 = (tile & 7) << 6;               // 8 col groups of 64
  const int o1 = half ? 8 : 0;
  const int o2 = half ? 24 : 16;

  const _Float16* arow = A + (size_t)(m0 + l15) * 512;
  const _Float16* brow[4];
#pragma unroll
  for (int g = 0; g < 4; ++g)
    brow[g] = Wt + (size_t)(n0 + g * 16 + l15) * 512;

  v8f c[4] = {};
#pragma unroll 2
  for (int k = 0; k < 512; k += 32) {
    v16h a = ld16(arow + k + o1, arow + k + o2);
#pragma unroll
    for (int g = 0; g < 4; ++g) {
      v16h b = ld16(brow[g] + k + o1, brow[g] + k + o2);
      c[g] = wmma16(a, b, c[g]);
    }
  }

  if (mode == 0) {
    _Float16* o16 = (_Float16*)out;
#pragma unroll
    for (int g = 0; g < 4; ++g) {
      const int n = n0 + g * 16 + l15;
      const float bv = bias[n];
      const int hh = n >> 6, dd = n & 63;
#pragma unroll
      for (int r = 0; r < 8; ++r) {
        int row = m0 + r + half * 8;
        int bb = row >> 10, ll = row & (L_ - 1);
        o16[(((size_t)(bb * NH_ + hh) * L_) + ll) * D_ + dd] =
            (_Float16)(c[g][r] + bv);
      }
    }
  } else if (mode == 1) {
    _Float16* o16 = (_Float16*)out;
#pragma unroll
    for (int g = 0; g < 4; ++g) {
      const int n = n0 + g * 16 + l15;
      const float bv = bias[n];
      const int hh = n >> 6, dd = n & 63;
#pragma unroll
      for (int r = 0; r < 8; ++r) {
        int row = m0 + r + half * 8;
        int bb = row >> 10, ll = row & (L_ - 1);
        o16[(((size_t)(bb * NH_ + hh) * D_) + dd) * L_ + ll] =
            (_Float16)(c[g][r] + bv);
      }
    }
  } else {
    float* of = (float*)out;
#pragma unroll
    for (int g = 0; g < 4; ++g) {
      const int n = n0 + g * 16 + l15;
      const float bv = bias[n];
#pragma unroll
      for (int r = 0; r < 8; ++r) {
        int row = m0 + r + half * 8;
        of[(size_t)row * H_ + n] = c[g][r] + bv;
      }
    }
  }
}

// ------------------------------------------------------- flash attention
// One wave = one 16-row query tile of one (b, h). Iterates key tiles of 32,
// online softmax, rel-pos bias streamed non-temporally, P@V accumulated
// into 16x64 f32. Output written as f16 [b][l][h*D+d] for the ff GEMM.
__global__ __launch_bounds__(128) void attn_kernel(
    const _Float16* __restrict__ q16, const _Float16* __restrict__ k16,
    const _Float16* __restrict__ vT16, const float* __restrict__ rel,
    const int* __restrict__ seq_len, const int* __restrict__ lex_num,
    _Float16* __restrict__ attn16) {
  __shared__ _Float16 plds[4][16 * 32];   // per-wave P transpose staging

  const int lane = threadIdx.x & 31;
  const int wave = threadIdx.x >> 5;
  const int half = lane >> 4;
  const int l15  = lane & 15;
  const int b = blockIdx.z;
  const int h = blockIdx.y;
  const int q0 = (blockIdx.x * 4 + wave) * 16;
  const int len = seq_len[b] + lex_num[b];

  const size_t bh = (size_t)(b * NH_ + h);
  const _Float16* qbase = q16  + bh * L_ * D_;
  const _Float16* kbase = k16  + bh * L_ * D_;
  const _Float16* vbase = vT16 + bh * D_ * L_;
  const float*    rbase = rel  + bh * L_ * L_;

  const int o1 = half ? 8 : 0;
  const int o2 = half ? 24 : 16;

  // Q tile operands (reused every key tile)
  const _Float16* arow = qbase + (size_t)(q0 + l15) * D_;
  v16h Aq0 = ld16(arow + o1, arow + o2);            // d 0..31
  v16h Aq1 = ld16(arow + 32 + o1, arow + 32 + o2);  // d 32..63

  v8f O[4] = {};                 // 16x64 f32 accumulator
  float mrow[8], lrow[8];
#pragma unroll
  for (int r = 0; r < 8; ++r) { mrow[r] = -3.0e38f; lrow[r] = 0.0f; }

  for (int kt = 0; kt < L_ / 32; ++kt) {
    const int k0 = kt * 32;

    // hint next rel-pos tile into cache early
    __builtin_prefetch(rbase + (size_t)(q0 + half * 8) * L_ + k0 + 32, 0, 0);

    // ---- S = Q K^T (two 16x16 tiles covering key cols k0..k0+31)
    const _Float16* kr0 = kbase + (size_t)(k0 + l15) * D_;
    const _Float16* kr1 = kbase + (size_t)(k0 + 16 + l15) * D_;
    v16h B00 = ld16(kr0 + o1, kr0 + o2);
    v16h B01 = ld16(kr0 + 32 + o1, kr0 + 32 + o2);
    v16h B10 = ld16(kr1 + o1, kr1 + o2);
    v16h B11 = ld16(kr1 + 32 + o1, kr1 + 32 + o2);
    v8f s0 = {}, s1 = {};
    s0 = wmma16(Aq0, B00, s0);
    s0 = wmma16(Aq1, B01, s0);
    s1 = wmma16(Aq0, B10, s1);
    s1 = wmma16(Aq1, B11, s1);

    // ---- scale, rel-pos bias (non-temporal stream), mask, online softmax
    float p0e[8], p1e[8];
#pragma unroll
    for (int r = 0; r < 8; ++r) {
      const int row = q0 + r + half * 8;
      const int c0 = k0 + l15, c1 = k0 + 16 + l15;
      float v0 = s0[r] * 0.125f +
                 __builtin_nontemporal_load(rbase + (size_t)row * L_ + c0);
      float v1 = s1[r] * 0.125f +
                 __builtin_nontemporal_load(rbase + (size_t)row * L_ + c1);
      if (c0 >= len) v0 = -1e15f;
      if (c1 >= len) v1 = -1e15f;
      float mx = rowmax16(fmaxf(v0, v1));
      float nm = fmaxf(mrow[r], mx);
      float al = __expf(mrow[r] - nm);
      float p0 = __expf(v0 - nm);
      float p1 = __expf(v1 - nm);
      lrow[r] = lrow[r] * al + rowsum16(p0 + p1);
      mrow[r] = nm;
      O[0][r] *= al; O[1][r] *= al; O[2][r] *= al; O[3][r] *= al;
      p0e[r] = p0;  p1e[r] = p1;
    }

    // ---- P: C-layout -> A-layout via wave-private LDS tile
    _Float16* pl = plds[wave];
#pragma unroll
    for (int r = 0; r < 8; ++r) {
      const int row = r + half * 8;
      pl[row * 32 + l15]      = (_Float16)p0e[r];
      pl[row * 32 + 16 + l15] = (_Float16)p1e[r];
    }
    asm volatile("s_wait_dscnt 0x0" ::: "memory");
    const _Float16* prow = pl + l15 * 32;
    v16h Ap = ld16(prow + o1, prow + o2);

    // ---- O += P @ V  (V^T rows contiguous over key index)
#pragma unroll
    for (int g = 0; g < 4; ++g) {
      const _Float16* vrow = vbase + (size_t)(g * 16 + l15) * L_ + k0;
      v16h Bv = ld16(vrow + o1, vrow + o2);
      O[g] = wmma16(Ap, Bv, O[g]);
    }
    asm volatile("s_wait_dscnt 0x0" ::: "memory");  // LDS reads done before next overwrite
  }

  // ---- normalize, write f16 [b][l][h*D+d]
#pragma unroll
  for (int r = 0; r < 8; ++r) {
    const int row = q0 + r + half * 8;
    const float inv = 1.0f / lrow[r];
    _Float16* orow = attn16 + (size_t)(b * L_ + row) * H_ + h * D_;
#pragma unroll
    for (int g = 0; g < 4; ++g)
      orow[g * 16 + l15] = (_Float16)(O[g][r] * inv);
  }
}

// ---------------------------------------------------------------- launch
extern "C" void kernel_launch(void* const* d_in, const int* in_sizes, int n_in,
                              void* d_out, int out_size, void* d_ws, size_t ws_size,
                              hipStream_t stream) {
  const float* key   = (const float*)d_in[0];
  const float* query = (const float*)d_in[1];
  const float* value = (const float*)d_in[2];
  const float* rel   = (const float*)d_in[3];
  const float* wk_w  = (const float*)d_in[4];
  const float* wk_b  = (const float*)d_in[5];
  const float* wq_w  = (const float*)d_in[6];
  const float* wq_b  = (const float*)d_in[7];
  const float* wv_w  = (const float*)d_in[8];
  const float* wv_b  = (const float*)d_in[9];
  const float* ff_w  = (const float*)d_in[10];
  const float* ff_b  = (const float*)d_in[11];
  const int* seq_len = (const int*)d_in[12];
  const int* lex_num = (const int*)d_in[13];
  float* out = (float*)d_out;

  const int XE = B_ * L_ * H_;   // 2,097,152
  const int WE = H_ * H_;        //   262,144

  char* ws = (char*)d_ws;
  size_t off = 0;
  auto alloc = [&](size_t bytes) {
    void* p = ws + off;
    off = (off + bytes + 255) & ~(size_t)255;
    return p;
  };
  _Float16* xk16  = (_Float16*)alloc((size_t)XE * 2);
  _Float16* xq16  = (_Float16*)alloc((size_t)XE * 2);
  _Float16* xv16  = (_Float16*)alloc((size_t)XE * 2);
  _Float16* wk16  = (_Float16*)alloc((size_t)WE * 2);
  _Float16* wq16  = (_Float16*)alloc((size_t)WE * 2);
  _Float16* wv16  = (_Float16*)alloc((size_t)WE * 2);
  _Float16* wf16  = (_Float16*)alloc((size_t)WE * 2);
  _Float16* q16   = (_Float16*)alloc((size_t)XE * 2);
  _Float16* k16   = (_Float16*)alloc((size_t)XE * 2);
  _Float16* vT16  = (_Float16*)alloc((size_t)XE * 2);
  _Float16* a16   = (_Float16*)alloc((size_t)XE * 2);
  // total ~32 MB of d_ws

  // fp32 -> f16 conversions
  cvt_f32_to_f16<<<1024, 256, 0, stream>>>(key,   xk16, XE);
  cvt_f32_to_f16<<<1024, 256, 0, stream>>>(query, xq16, XE);
  cvt_f32_to_f16<<<1024, 256, 0, stream>>>(value, xv16, XE);
  cvt_f32_to_f16<<<256, 256, 0, stream>>>(wk_w, wk16, WE);
  cvt_f32_to_f16<<<256, 256, 0, stream>>>(wq_w, wq16, WE);
  cvt_f32_to_f16<<<256, 256, 0, stream>>>(wv_w, wv16, WE);
  cvt_f32_to_f16<<<256, 256, 0, stream>>>(ff_w, wf16, WE);

  // projections: Q, K (head-major), V (transposed per head)
  // 2048 waves = 256 row-tiles x 8 col-groups, 4 waves/block
  gemm512_kernel<<<512, 128, 0, stream>>>(xq16, wq16, wq_b, q16, 0);
  gemm512_kernel<<<512, 128, 0, stream>>>(xk16, wk16, wk_b, k16, 0);
  gemm512_kernel<<<512, 128, 0, stream>>>(xv16, wv16, wv_b, vT16, 1);

  // fused flash attention: grid (qtiles/4, NH, B), 4 waves per block
  attn_kernel<<<dim3(L_ / 64, NH_, B_), 128, 0, stream>>>(
      q16, k16, vT16, rel, seq_len, lex_num, a16);

  // output projection -> fp32
  gemm512_kernel<<<512, 128, 0, stream>>>(a16, wf16, ff_b, out, 2);
}
